// Encoder_overall_78700980731992
// MI455X (gfx1250) — matrix operand order repaired and checked by reading, not compiled, not run on hardware
//
#include <hip/hip_runtime.h>
#include <hip/hip_bf16.h>
#include <stdint.h>

// ---------------------------------------------------------------------------
// MI455X (gfx1250) implementation of the multi-omics encoder.
// Heavy matmuls -> v_wmma_f32_16x16x32_bf16 (f32 accumulate); A-tiles are
// streamed into LDS by the Tensor Data Mover (tensor_load_to_lds +
// s_wait_tensorcnt); attention is flash-style; global_prefetch_b8 pulls the
// next K-tiles while WMMAs execute.
// ---------------------------------------------------------------------------

typedef __bf16 bf16;
typedef __attribute__((ext_vector_type(16))) __bf16 bf16x16;
typedef __attribute__((ext_vector_type(8)))  float  floatx8;
typedef __attribute__((ext_vector_type(4)))  unsigned int v4u;
typedef __attribute__((ext_vector_type(8)))  int v8i;
typedef __attribute__((ext_vector_type(4)))  int v4i;

#define N_SPOTS 4096
#define DIM     64
#define DIN1_C  1024
#define DIN2_C  256
#define NEDGE   512

union AFrag { bf16x16 v; unsigned int u[8]; };

// A-fragment (16x32 bf16): lane = 16*half + m. VGPR j holds K pair:
//   j<4 : K = 2j   + 8*half      -> dword index j + 4*half
//   j>=4: K = 16+2(j-4) + 8*half -> dword index 8 + (j-4) + 4*half
__device__ __forceinline__ int mapA(int j, int half) {
  return (j < 4) ? (j + 4 * half) : (4 + j + 4 * half);
}
// B-fragment (32x16 bf16): lane = 16*half + n. VGPR j holds the same K pair as mapA.
__device__ __forceinline__ int mapB(int j, int half) { return (j < 4) ? (j + 4 * half) : (4 + j + 4 * half); }

// ---------------------------------------------------------------------------
// TDM: issue a 2-D tensor_load_to_lds of a (tile_d1 rows x tile_d0 elems)
// bf16 tile starting at gptr (row stride = stride_elems) into LDS lds_off.
// D# packing per CDNA5 ISA.
// ---------------------------------------------------------------------------
__device__ __forceinline__ void tdm_load_tile_2d(
    unsigned lds_off, const void* gptr, unsigned tile_d0, unsigned tile_d1,
    unsigned long long stride_elems, unsigned tensor_d0, unsigned tensor_d1) {
  const unsigned long long ga = (unsigned long long)(uintptr_t)gptr;
  v4u g0;
  g0.x = 1u;                                                 // count=1
  g0.y = lds_off;                                            // lds_addr
  g0.z = (unsigned)ga;                                       // global_addr[31:0]
  g0.w = (unsigned)((ga >> 32) & 0x01ffffffu) | (2u << 30);  // [56:32] + type=2
  v8i g1;
  g1[0] = (int)(1u << 16);                                   // data_size=1 (2B)
  g1[1] = (int)((tensor_d0 & 0xffffu) << 16);                // tensor_dim0 lo16
  g1[2] = (int)((tensor_d0 >> 16) & 0xffffu)                 // tensor_dim0 hi16
        | (int)((tensor_d1 & 0xffffu) << 16);                // tensor_dim1 lo16
  g1[3] = (int)((tensor_d1 >> 16) & 0xffffu)                 // tensor_dim1 hi16
        | (int)(tile_d0 << 16);                              // tile_dim0
  g1[4] = (int)tile_d1;                                      // tile_dim1 (dim2=0)
  g1[5] = (int)(unsigned)stride_elems;                       // dim0 stride lo32
  g1[6] = (int)((stride_elems >> 32) & 0xffffu);             // dim0 stride hi16
  g1[7] = 0;                                                 // dim1 stride = 0
  v4i gz = {0, 0, 0, 0};
#if __clang_major__ >= 23
  v8i gz8 = {0, 0, 0, 0, 0, 0, 0, 0};
  __builtin_amdgcn_tensor_load_to_lds(g0, g1, gz, gz, gz8, 0);
#else
  __builtin_amdgcn_tensor_load_to_lds(g0, g1, gz, gz, 0);
#endif
}

// ---------------------------------------------------------------------------
// Generic bf16 GEMM: C[M,Nc] = act(A[M,K] @ B[K,Nc] + bias + addres)
// 128x64 macro tile per 256-thread block (8 waves); each wave owns a 16x64
// strip -> 4 back-to-back WMMAs per k-step with one shared A-fragment.
// A tile arrives via TDM; B tile is transpose-staged as packed K-pair dwords.
// ---------------------------------------------------------------------------
__global__ __launch_bounds__(256)
void gemm_bf16(const bf16* __restrict__ A, const bf16* __restrict__ B,
               const float* __restrict__ bias, const float* __restrict__ addres,
               float* __restrict__ outF, bf16* __restrict__ outB,
               int M, int Nc, int K, int ldb, int ldc, int relu)
{
  __shared__ __align__(16) bf16         As[128][32];   // 8 KB (TDM dest)
  __shared__ __align__(16) unsigned int Bs32[64][16];  // 4 KB: [n][kpair]
  const int t = threadIdx.x;
  const int w = t >> 5, lane = t & 31, half = lane >> 4, mrow = lane & 15;
  const int Mb = blockIdx.y * 128, Nb = blockIdx.x * 64;
  floatx8 acc[4] = {{}, {}, {}, {}};

  const int kp = t >> 4, ng4 = (t & 15) * 4;     // B stage: 16 k-pairs x 64 n
  const unsigned asBase = (unsigned)(uintptr_t)(void*)&As[0][0];

  for (int k0 = 0; k0 < K; k0 += 32) {
    // A tile 128x32 bf16 via Tensor Data Mover (one wave issues the DMA)
    if (t < 32)
      tdm_load_tile_2d(asBase, A + (size_t)Mb * K + k0, 32u, 128u,
                       (unsigned long long)K, (unsigned)K, (unsigned)M);
    // stage B transposed as packed K-pairs: 4 x ds_store_b32 per thread
    union { uint2 q; unsigned short s[4]; } r0, r1;
    r0.q = *(const uint2*)(B + (size_t)(k0 + 2 * kp) * ldb + Nb + ng4);
    r1.q = *(const uint2*)(B + (size_t)(k0 + 2 * kp + 1) * ldb + Nb + ng4);
#pragma unroll
    for (int i = 0; i < 4; ++i)
      Bs32[ng4 + i][kp] = (unsigned int)r0.s[i] | ((unsigned int)r1.s[i] << 16);
    // speculative prefetch of the next K tile (global_prefetch_b8)
    if (k0 + 32 < K) {
      __builtin_prefetch(A + (size_t)(Mb + (t >> 1)) * K + k0 + 32, 0, 1);
      __builtin_prefetch(B + (size_t)(k0 + 32 + 2 * kp) * ldb + Nb + ng4, 0, 1);
    }
    if (t < 32) __builtin_amdgcn_s_wait_tensorcnt(0);  // TDM tile landed
    __syncthreads();

    // gather all fragments first so the 4 WMMAs can issue back-to-back
    AFrag a;
    const unsigned int* ar = (const unsigned int*)&As[w * 16 + mrow][0];
#pragma unroll
    for (int j = 0; j < 8; ++j) a.u[j] = ar[mapA(j, half)];
    AFrag bfr[4];
#pragma unroll
    for (int ti = 0; ti < 4; ++ti) {
      const unsigned int* br = &Bs32[ti * 16 + mrow][0];
#pragma unroll
      for (int j = 0; j < 8; ++j) bfr[ti].u[j] = br[mapB(j, half)];
    }
#pragma unroll
    for (int ti = 0; ti < 4; ++ti)
      acc[ti] = __builtin_amdgcn_wmma_f32_16x16x32_bf16(
          false, a.v, false, bfr[ti].v, (short)0, acc[ti], false, false);
    __syncthreads();
  }

  const int r0r = Mb + w * 16 + 8 * half;
  const int c0 = Nb + mrow;
#pragma unroll
  for (int v = 0; v < 8; ++v) {
    const int r = r0r + v;
#pragma unroll
    for (int ti = 0; ti < 4; ++ti) {
      const int c = c0 + ti * 16;
      float val = acc[ti][v];
      if (bias)   val += bias[c];
      if (addres) val += addres[(size_t)r * ldc + c];
      if (relu)   val = fmaxf(val, 0.f);
      if (outF) outF[(size_t)r * ldc + c] = val;
      if (outB) outB[(size_t)r * ldc + c] = (bf16)val;
    }
  }
}

// ---------------------------------------------------------------------------
// Flash attention: out = softmax(mask(Q K^T * scale)) @ V, streamed keys.
// Each wave owns 16 query rows. HD = 32 (encoder heads) or 64 (cross/causal).
// V supplied transposed (Vt[d][key]) so PV B-fragments are contiguous reads.
// ---------------------------------------------------------------------------
template <int HD>
__global__ __launch_bounds__(128)
void flash_attn(const bf16* __restrict__ Q, const bf16* __restrict__ K,
                const bf16* __restrict__ Vt, const int* __restrict__ mask,
                float* __restrict__ out, int n, int ldq, int ldk, int ldvt,
                int ldo, float scale)
{
  constexpr int KB = HD / 32;   // k-chunks for QK^T
  constexpr int DT = HD / 16;   // output d-tiles
  __shared__ __align__(16) bf16 Pl[4][16][32];   // per-wave P staging
  const int t = threadIdx.x, w = t >> 5, lane = t & 31;
  const int half = lane >> 4, mrow = lane & 15;
  const int qbase = blockIdx.x * 64 + w * 16;

  AFrag aq[KB];
#pragma unroll
  for (int c = 0; c < KB; ++c) {
    const unsigned int* qr =
        (const unsigned int*)(Q + (size_t)(qbase + mrow) * ldq + c * 32);
#pragma unroll
    for (int j = 0; j < 8; ++j) aq[c].u[j] = qr[mapA(j, half)];
  }

  float mI[8], lI[8];
#pragma unroll
  for (int v = 0; v < 8; ++v) { mI[v] = -1e30f; lI[v] = 0.f; }
  floatx8 acc[DT];
#pragma unroll
  for (int d = 0; d < DT; ++d) acc[d] = {};

  for (int j0 = 0; j0 < n; j0 += 32) {
    if (j0 + 32 < n) {  // prefetch next key chunk (global_prefetch_b8)
      __builtin_prefetch(K + (size_t)(j0 + 32 + mrow) * ldk, 0, 1);
      __builtin_prefetch(Vt + (size_t)mrow * ldvt + j0 + 32, 0, 1);
      if (mask) __builtin_prefetch(mask + (size_t)(qbase + mrow) * n + j0 + 32, 0, 1);
    }
    floatx8 s0 = {}, s1 = {};
#pragma unroll
    for (int c = 0; c < KB; ++c) {
      AFrag b0, b1;
      const unsigned int* kr0 =
          (const unsigned int*)(K + (size_t)(j0 + mrow) * ldk + c * 32);
      const unsigned int* kr1 =
          (const unsigned int*)(K + (size_t)(j0 + 16 + mrow) * ldk + c * 32);
#pragma unroll
      for (int j = 0; j < 8; ++j) {
        b0.u[j] = kr0[mapB(j, half)];
        b1.u[j] = kr1[mapB(j, half)];
      }
      s0 = __builtin_amdgcn_wmma_f32_16x16x32_bf16(false, aq[c].v, false, b0.v,
                                                   (short)0, s0, false, false);
      s1 = __builtin_amdgcn_wmma_f32_16x16x32_bf16(false, aq[c].v, false, b1.v,
                                                   (short)0, s1, false, false);
    }
    // online softmax over the 32 keys of this chunk
#pragma unroll
    for (int v = 0; v < 8; ++v) {
      const int qr = qbase + v + 8 * half;
      float x0 = s0[v] * scale, x1 = s1[v] * scale;
      if (mask) {
        if (mask[(size_t)qr * n + j0 + mrow] == 0)      x0 = -1e30f;
        if (mask[(size_t)qr * n + j0 + 16 + mrow] == 0) x1 = -1e30f;
      }
      float mx = fmaxf(x0, x1);
#pragma unroll
      for (int d = 1; d <= 8; d <<= 1) mx = fmaxf(mx, __shfl_xor(mx, d, 32));
      const float mN = fmaxf(mI[v], mx);
      const float al = __expf(mI[v] - mN);
      const float p0 = __expf(x0 - mN), p1 = __expf(x1 - mN);
      float rs = p0 + p1;
#pragma unroll
      for (int d = 1; d <= 8; d <<= 1) rs += __shfl_xor(rs, d, 32);
      lI[v] = lI[v] * al + rs;
      mI[v] = mN;
#pragma unroll
      for (int dt = 0; dt < DT; ++dt) acc[dt][v] *= al;
      Pl[w][v + 8 * half][mrow]      = (bf16)p0;
      Pl[w][v + 8 * half][16 + mrow] = (bf16)p1;
    }
    // P (16x32) back as an A-fragment, then PV (fragments hoisted so the
    // DT WMMAs issue back-to-back)
    AFrag pa;
    const unsigned int* pr = (const unsigned int*)&Pl[w][mrow][0];
#pragma unroll
    for (int j = 0; j < 8; ++j) pa.u[j] = pr[mapA(j, half)];
    AFrag bv[DT];
#pragma unroll
    for (int dt = 0; dt < DT; ++dt) {
      const unsigned int* vr =
          (const unsigned int*)(Vt + (size_t)(dt * 16 + mrow) * ldvt + j0);
#pragma unroll
      for (int j = 0; j < 8; ++j) bv[dt].u[j] = vr[mapB(j, half)];
    }
#pragma unroll
    for (int dt = 0; dt < DT; ++dt)
      acc[dt] = __builtin_amdgcn_wmma_f32_16x16x32_bf16(
          false, pa.v, false, bv[dt].v, (short)0, acc[dt], false, false);
  }
#pragma unroll
  for (int v = 0; v < 8; ++v) {
    const int qr = qbase + v + 8 * half;
    const float inv = 1.f / lI[v];
#pragma unroll
    for (int dt = 0; dt < DT; ++dt)
      out[(size_t)qr * ldo + dt * 16 + mrow] = acc[dt][v] * inv;
  }
}

// ---------------------------------------------------------------------------
// Small utility kernels
// ---------------------------------------------------------------------------
__global__ void cvt_bf16(const float* __restrict__ in, bf16* __restrict__ out,
                         size_t n) {  // n must be even; packed b32 stores
  size_t i = ((size_t)blockIdx.x * blockDim.x + threadIdx.x) * 2;
  const size_t stride = (size_t)gridDim.x * blockDim.x * 2;
  for (; i < n; i += stride) {
    union { unsigned int u; bf16 h[2]; } p;
    p.h[0] = (bf16)in[i];
    p.h[1] = (bf16)in[i + 1];
    *(unsigned int*)(out + i) = p.u;
  }
}

__global__ __launch_bounds__(256)
void transpose_cvt(const float* __restrict__ in, bf16* __restrict__ out,
                   int R, int C) {  // out[c*R+r] = in[r*C+c]
  __shared__ float tile[32][33];
  const int cb = blockIdx.x * 32, rb = blockIdx.y * 32;
  const int tx = threadIdx.x & 31, ty = threadIdx.x >> 5;
#pragma unroll
  for (int i = 0; i < 4; ++i)
    tile[ty + i * 8][tx] = in[(size_t)(rb + ty + i * 8) * C + cb + tx];
  __syncthreads();
#pragma unroll
  for (int i = 0; i < 4; ++i)
    out[(size_t)(cb + ty + i * 8) * R + rb + tx] = (bf16)tile[tx][ty + i * 8];
}

__global__ __launch_bounds__(256)
void ln_rows(const float* __restrict__ x, const float* __restrict__ res,
             const float* __restrict__ g, const float* __restrict__ b,
             float* __restrict__ outF, bf16* __restrict__ outB) {
  const int row = blockIdx.x * 8 + (threadIdx.x >> 5);
  const int lane = threadIdx.x & 31;
  const size_t base = (size_t)row * 64;
  float a0 = x[base + lane], a1 = x[base + lane + 32];
  if (res) { a0 += res[base + lane]; a1 += res[base + lane + 32]; }
  float s = a0 + a1;
#pragma unroll
  for (int d = 1; d <= 16; d <<= 1) s += __shfl_xor(s, d, 32);
  const float mean = s * (1.f / 64.f);
  const float d0 = a0 - mean, d1 = a1 - mean;
  float vv = d0 * d0 + d1 * d1;
#pragma unroll
  for (int d = 1; d <= 16; d <<= 1) vv += __shfl_xor(vv, d, 32);
  const float rstd = rsqrtf(vv * (1.f / 64.f) + 1e-5f);
  const float y0 = d0 * rstd * g[lane] + b[lane];
  const float y1 = d1 * rstd * g[lane + 32] + b[lane + 32];
  if (outF) { outF[base + lane] = y0; outF[base + lane + 32] = y1; }
  if (outB) { outB[base + lane] = (bf16)y0; outB[base + lane + 32] = (bf16)y1; }
}

__global__ __launch_bounds__(256)
void spatial_ln(const float* __restrict__ p, const float* __restrict__ coords,
                const float* __restrict__ cw, const float* __restrict__ cb,
                const float* __restrict__ g, const float* __restrict__ b,
                float* __restrict__ outF, bf16* __restrict__ outB) {
  const int row = blockIdx.x * 8 + (threadIdx.x >> 5);
  const int lane = threadIdx.x & 31;
  const size_t base = (size_t)row * 64;
  const float c0 = coords[row * 2 + 0], c1 = coords[row * 2 + 1];
  const float t0 = fmaxf(c0 * cw[lane] + c1 * cw[64 + lane] + cb[lane], 0.f);
  const float t1 = fmaxf(c0 * cw[lane + 32] + c1 * cw[96 + lane] + cb[lane + 32], 0.f);
  float a0 = p[base + lane] + t0;        // p already relu(feat@fw+fb)
  float a1 = p[base + lane + 32] + t1;
  float s = a0 + a1;
#pragma unroll
  for (int d = 1; d <= 16; d <<= 1) s += __shfl_xor(s, d, 32);
  const float mean = s * (1.f / 64.f);
  const float d0 = a0 - mean, d1 = a1 - mean;
  float vv = d0 * d0 + d1 * d1;
#pragma unroll
  for (int d = 1; d <= 16; d <<= 1) vv += __shfl_xor(vv, d, 32);
  const float rstd = rsqrtf(vv * (1.f / 64.f) + 1e-5f);
  const float y0 = d0 * rstd * g[lane] + b[lane];
  const float y1 = d1 * rstd * g[lane + 32] + b[lane + 32];
  outF[base + lane] = y0; outF[base + lane + 32] = y1;
  outB[base + lane] = (bf16)y0; outB[base + lane + 32] = (bf16)y1;
}

__global__ __launch_bounds__(256)
void atten_fuse(const float* __restrict__ t0, const float* __restrict__ t1,
                const float* __restrict__ u, const float* __restrict__ es,
                const float* __restrict__ ef, float* __restrict__ outF,
                bf16* __restrict__ outB) {
  const int row = blockIdx.x * 8 + (threadIdx.x >> 5);
  const int lane = threadIdx.x & 31;
  const size_t base = (size_t)row * 64;
  float s0 = tanhf(t0[base + lane]) * u[lane] + tanhf(t0[base + lane + 32]) * u[lane + 32];
  float s1 = tanhf(t1[base + lane]) * u[lane] + tanhf(t1[base + lane + 32]) * u[lane + 32];
#pragma unroll
  for (int d = 1; d <= 16; d <<= 1) {
    s0 += __shfl_xor(s0, d, 32);
    s1 += __shfl_xor(s1, d, 32);
  }
  const float mmx = fmaxf(s0, s1);
  const float e0 = __expf(s0 - mmx), e1 = __expf(s1 - mmx);
  const float a0 = e0 / (e0 + e1), a1 = 1.f - a0;
  const float y0 = a0 * es[base + lane] + a1 * ef[base + lane];
  const float y1 = a0 * es[base + lane + 32] + a1 * ef[base + lane + 32];
  outF[base + lane] = y0; outF[base + lane + 32] = y1;
  outB[base + lane] = (bf16)y0; outB[base + lane + 32] = (bf16)y1;
}

__global__ void dv_kernel(const float* __restrict__ H, float* __restrict__ dvis) {
  const int r = blockIdx.x * blockDim.x + threadIdx.x;
  if (r >= N_SPOTS) return;
  float s = 0.f;
  const float* row = H + (size_t)r * NEDGE;
  for (int e = 0; e < NEDGE; ++e) s += row[e];
  dvis[r] = 1.f / (sqrtf(s) + 1e-6f);
}

__global__ void de_kernel(const float* __restrict__ H, float* __restrict__ deinv) {
  const int e = blockIdx.x * blockDim.x + threadIdx.x;
  if (e >= NEDGE) return;
  float s = 0.f;
  for (int r = 0; r < N_SPOTS; ++r) s += H[(size_t)r * NEDGE + e];
  deinv[e] = 1.f / (s + 1e-6f);
}

__global__ void rowscale_cvt(const float* __restrict__ x, const float* __restrict__ sc,
                             bf16* __restrict__ out, int n) {
  const int i = blockIdx.x * blockDim.x + threadIdx.x;
  if (i < n) out[i] = (bf16)(x[i] * sc[i >> 6]);
}

__global__ void final1_k(const float* __restrict__ e1, const float* __restrict__ caus,
                         const float* __restrict__ t2, const float* __restrict__ dvis,
                         bf16* __restrict__ out, int n) {
  const int i = blockIdx.x * blockDim.x + threadIdx.x;
  if (i < n) out[i] = (bf16)(e1[i] + caus[i] + fmaxf(dvis[i >> 6] * t2[i], 0.f));
}

__global__ void final2_k(const float* __restrict__ e2, const float* __restrict__ att,
                         bf16* __restrict__ out, int n) {
  const int i = blockIdx.x * blockDim.x + threadIdx.x;
  if (i < n) out[i] = (bf16)(e2[i] + att[i]);
}

// ---------------------------------------------------------------------------
// Host orchestration
// ---------------------------------------------------------------------------
// Input index map: 8 data arrays, then params pytree alphabetically flattened.
enum {
  I_F1 = 0, I_F2, I_COORD, I_ADJ, I_H, I_ADJF1, I_ADJF2, I_CMASK,
  I_ATT1_U = 8, I_ATT1_W, I_ATT2_U, I_ATT2_W,
  I_CAUS_KB = 12, I_CAUS_KW, I_CAUS_QB, I_CAUS_QW, I_CAUS_VB, I_CAUS_VW,
  I_CROSS_KW = 18, I_CROSS_QW, I_CROSS_VW,
  I_DEC1 = 21, I_DEC2 = 22,
  I_E1_BASE = 23,  // f1b,f1w,f2b,f2w,kb,kw,ln_b,ln_g,ob,ow,qb,qw,rb,rw,vb,vw
  I_E2_BASE = 39,
  I_HG = 55,
  I_SP1_BASE = 56, // cb,cw,fb,fw,gw,ln_b,ln_g
  I_SP2_BASE = 63
};

// Workspace offsets in 512KB units (total ~101 MB).
enum {
  O_ADJBF = 0, O_F1BF = 64, O_F2BF = 80, O_HBF = 84, O_HTBF = 92, O_WPOOL = 100,
  O_QBF = 108, O_KBF = 109, O_VF = 110, O_VTBF = 112, O_RES = 113, O_HOUT = 115,
  O_HOUTBF = 117, O_HO = 118, O_H1 = 120, O_H1BF = 122, O_FFBF = 123, O_FF2 = 124,
  O_EF1 = 126, O_EF1BF = 128, O_EF2 = 129, O_EF2BF = 131, O_SPTMP = 132, O_SPX = 134,
  O_SPXBF = 136, O_SPA = 137, O_SPABF = 139, O_ES1 = 140, O_ES1BF = 142, O_ES2 = 143,
  O_ES2BF = 145, O_AT0 = 146, O_AT1 = 148, O_E1 = 150, O_E1BF = 152, O_E2 = 153,
  O_E2BF = 155, O_CQBF = 156, O_CKBF = 157, O_CVF = 158, O_CVTBF = 160, O_CATT = 161,
  O_XQBF = 163, O_XKBF = 164, O_XVF = 165, O_XVTBF = 167, O_CAUS = 168, O_X = 170,
  O_XSBF = 172, O_DV = 173, O_DE = 174, O_T1 = 175, O_T1SBF = 176, O_T2 = 177,
  O_FIN1BF = 179, O_FIN2BF = 180, O_Y1BF = 181, O_Y2BF = 197
};

extern "C" void kernel_launch(void* const* d_in, const int* in_sizes, int n_in,
                              void* d_out, int out_size, void* d_ws, size_t ws_size,
                              hipStream_t stream) {
  (void)in_sizes; (void)n_in; (void)out_size; (void)ws_size;
  char* ws = (char*)d_ws;
  auto F   = [&](int i) { return (const float*)d_in[i]; };
  auto Ii  = [&](int i) { return (const int*)d_in[i]; };
  auto WSB = [&](int u) { return (bf16*)(ws + (size_t)u * (512 * 1024)); };
  auto WSF = [&](int u) { return (float*)(ws + (size_t)u * (512 * 1024)); };

  auto cvt = [&](const float* src, bf16* dst, size_t n) {
    int blocks = (int)((n + 4095) / 4096);
    if (blocks > 4096) blocks = 4096;
    if (blocks < 1) blocks = 1;
    cvt_bf16<<<blocks, 256, 0, stream>>>(src, dst, n);
  };
  auto gemm = [&](const bf16* A, const bf16* B, const float* bias,
                  const float* addres, float* oF, bf16* oB, int M, int Nc,
                  int K, int ldb, int ldc, int relu) {
    gemm_bf16<<<dim3(Nc / 64, M / 128), 256, 0, stream>>>(
        A, B, bias, addres, oF, oB, M, Nc, K, ldb, ldc, relu);
  };

  // ---- one-time conversions to bf16 ---------------------------------------
  cvt(F(I_ADJ), WSB(O_ADJBF), (size_t)N_SPOTS * N_SPOTS);
  cvt(F(I_F1),  WSB(O_F1BF),  (size_t)N_SPOTS * DIN1_C);
  cvt(F(I_F2),  WSB(O_F2BF),  (size_t)N_SPOTS * DIN2_C);
  cvt(F(I_H),   WSB(O_HBF),   (size_t)N_SPOTS * NEDGE);
  transpose_cvt<<<dim3(NEDGE / 32, N_SPOTS / 32), 256, 0, stream>>>(
      F(I_H), WSB(O_HTBF), N_SPOTS, NEDGE);

  size_t woff = 0;
  auto cw = [&](int idx, size_t n) -> const bf16* {
    bf16* dst = WSB(O_WPOOL) + woff;
    cvt(F(idx), dst, n);
    woff += (n + 255) & ~(size_t)255;
    return dst;
  };

  // ---- graph transformer encoders -----------------------------------------
  struct Enc { int din; const bf16* xbf; const int* mask; int pb; float* ef; bf16* efbf; };
  Enc encs[2] = {
      { DIN1_C, WSB(O_F1BF), Ii(I_ADJF1), I_E1_BASE, WSF(O_EF1), WSB(O_EF1BF) },
      { DIN2_C, WSB(O_F2BF), Ii(I_ADJF2), I_E2_BASE, WSF(O_EF2), WSB(O_EF2BF) },
  };
  for (int e = 0; e < 2; ++e) {
    const Enc& E = encs[e];
    const bf16* Wr  = cw(E.pb + 13, (size_t)E.din * DIM);
    const bf16* Wq  = cw(E.pb + 11, (size_t)E.din * DIM);
    const bf16* Wk  = cw(E.pb + 5,  (size_t)E.din * DIM);
    const bf16* Wv  = cw(E.pb + 15, (size_t)E.din * DIM);
    const bf16* Wo  = cw(E.pb + 9,  DIM * DIM);
    const bf16* Wf1 = cw(E.pb + 1,  DIM * DIM);
    const bf16* Wf2 = cw(E.pb + 3,  DIM * DIM);
    const float* lng = F(E.pb + 7); const float* lnb = F(E.pb + 6);

    gemm(E.xbf, Wr, F(E.pb + 12), nullptr, WSF(O_RES), nullptr, N_SPOTS, DIM, E.din, DIM, DIM, 0);
    gemm(E.xbf, Wq, F(E.pb + 10), nullptr, nullptr, WSB(O_QBF), N_SPOTS, DIM, E.din, DIM, DIM, 0);
    gemm(E.xbf, Wk, F(E.pb + 4),  nullptr, nullptr, WSB(O_KBF), N_SPOTS, DIM, E.din, DIM, DIM, 0);
    gemm(E.xbf, Wv, F(E.pb + 14), nullptr, WSF(O_VF), nullptr, N_SPOTS, DIM, E.din, DIM, DIM, 0);
    transpose_cvt<<<dim3(DIM / 32, N_SPOTS / 32), 256, 0, stream>>>(
        WSF(O_VF), WSB(O_VTBF), N_SPOTS, DIM);
    for (int h = 0; h < 2; ++h)
      flash_attn<32><<<dim3(N_SPOTS / 64), 128, 0, stream>>>(
          WSB(O_QBF) + 32 * h, WSB(O_KBF) + 32 * h,
          WSB(O_VTBF) + (size_t)32 * h * N_SPOTS, E.mask,
          WSF(O_HOUT) + 32 * h, N_SPOTS, DIM, DIM, N_SPOTS, DIM,
          0.1767766952966369f /* 1/sqrt(32) */);
    cvt(WSF(O_HOUT), WSB(O_HOUTBF), (size_t)N_SPOTS * DIM);
    gemm(WSB(O_HOUTBF), Wo, F(E.pb + 8), nullptr, WSF(O_HO), nullptr, N_SPOTS, DIM, DIM, DIM, DIM, 0);
    ln_rows<<<N_SPOTS / 8, 256, 0, stream>>>(WSF(O_HO), WSF(O_RES), lng, lnb, WSF(O_H1), WSB(O_H1BF));
    gemm(WSB(O_H1BF), Wf1, F(E.pb + 0), nullptr, nullptr, WSB(O_FFBF), N_SPOTS, DIM, DIM, DIM, DIM, 1);
    gemm(WSB(O_FFBF), Wf2, F(E.pb + 2), nullptr, WSF(O_FF2), nullptr, N_SPOTS, DIM, DIM, DIM, DIM, 0);
    ln_rows<<<N_SPOTS / 8, 256, 0, stream>>>(WSF(O_FF2), WSF(O_H1), lng, lnb, E.ef, E.efbf);
  }

  // ---- spatial GCNs --------------------------------------------------------
  struct Sp { int din; const bf16* xbf; int pb; float* es; bf16* esbf; };
  Sp sps[2] = {
      { DIN1_C, WSB(O_F1BF), I_SP1_BASE, WSF(O_ES1), WSB(O_ES1BF) },
      { DIN2_C, WSB(O_F2BF), I_SP2_BASE, WSF(O_ES2), WSB(O_ES2BF) },
  };
  for (int s = 0; s < 2; ++s) {
    const Sp& S = sps[s];
    const bf16* Wf = cw(S.pb + 3, (size_t)S.din * DIM);
    const bf16* Wg = cw(S.pb + 4, DIM * DIM);
    gemm(S.xbf, Wf, F(S.pb + 2), nullptr, WSF(O_SPTMP), nullptr, N_SPOTS, DIM, S.din, DIM, DIM, 1);
    spatial_ln<<<N_SPOTS / 8, 256, 0, stream>>>(
        WSF(O_SPTMP), F(I_COORD), F(S.pb + 1), F(S.pb + 0), F(S.pb + 6), F(S.pb + 5),
        WSF(O_SPX), WSB(O_SPXBF));
    gemm(WSB(O_ADJBF), WSB(O_SPXBF), nullptr, nullptr, WSF(O_SPA), WSB(O_SPABF),
         N_SPOTS, DIM, N_SPOTS, DIM, DIM, 0);
    gemm(WSB(O_SPABF), Wg, nullptr, nullptr, S.es, S.esbf, N_SPOTS, DIM, DIM, DIM, DIM, 1);
  }

  // ---- attention gating (modality fusion) ---------------------------------
  {
    const bf16* Wa1 = cw(I_ATT1_W, DIM * DIM);
    gemm(WSB(O_ES1BF), Wa1, nullptr, nullptr, WSF(O_AT0), nullptr, N_SPOTS, DIM, DIM, DIM, DIM, 0);
    gemm(WSB(O_EF1BF), Wa1, nullptr, nullptr, WSF(O_AT1), nullptr, N_SPOTS, DIM, DIM, DIM, DIM, 0);
    atten_fuse<<<N_SPOTS / 8, 256, 0, stream>>>(
        WSF(O_AT0), WSF(O_AT1), F(I_ATT1_U), WSF(O_ES1), WSF(O_EF1), WSF(O_E1), WSB(O_E1BF));
    const bf16* Wa2 = cw(I_ATT2_W, DIM * DIM);
    gemm(WSB(O_ES2BF), Wa2, nullptr, nullptr, WSF(O_AT0), nullptr, N_SPOTS, DIM, DIM, DIM, DIM, 0);
    gemm(WSB(O_EF2BF), Wa2, nullptr, nullptr, WSF(O_AT1), nullptr, N_SPOTS, DIM, DIM, DIM, DIM, 0);
    atten_fuse<<<N_SPOTS / 8, 256, 0, stream>>>(
        WSF(O_AT0), WSF(O_AT1), F(I_ATT2_U), WSF(O_ES2), WSF(O_EF2), WSF(O_E2), WSB(O_E2BF));
  }

  // ---- cross-modal attention (no mask) ------------------------------------
  {
    const bf16* Wq = cw(I_CROSS_QW, DIM * DIM);
    const bf16* Wk = cw(I_CROSS_KW, DIM * DIM);
    const bf16* Wv = cw(I_CROSS_VW, DIM * DIM);
    gemm(WSB(O_E1BF), Wq, nullptr, nullptr, nullptr, WSB(O_CQBF), N_SPOTS, DIM, DIM, DIM, DIM, 0);
    gemm(WSB(O_E2BF), Wk, nullptr, nullptr, nullptr, WSB(O_CKBF), N_SPOTS, DIM, DIM, DIM, DIM, 0);
    gemm(WSB(O_E2BF), Wv, nullptr, nullptr, WSF(O_CVF), nullptr, N_SPOTS, DIM, DIM, DIM, DIM, 0);
    transpose_cvt<<<dim3(DIM / 32, N_SPOTS / 32), 256, 0, stream>>>(
        WSF(O_CVF), WSB(O_CVTBF), N_SPOTS, DIM);
    flash_attn<64><<<dim3(N_SPOTS / 64), 128, 0, stream>>>(
        WSB(O_CQBF), WSB(O_CKBF), WSB(O_CVTBF), nullptr, WSF(O_CATT),
        N_SPOTS, DIM, DIM, N_SPOTS, DIM, 0.125f);
  }

  // ---- causal attention (masked) ------------------------------------------
  {
    const bf16* Wq = cw(I_CAUS_QW, DIM * DIM);
    const bf16* Wk = cw(I_CAUS_KW, DIM * DIM);
    const bf16* Wv = cw(I_CAUS_VW, DIM * DIM);
    gemm(WSB(O_E1BF), Wq, F(I_CAUS_QB), nullptr, nullptr, WSB(O_XQBF), N_SPOTS, DIM, DIM, DIM, DIM, 0);
    gemm(WSB(O_E2BF), Wk, F(I_CAUS_KB), nullptr, nullptr, WSB(O_XKBF), N_SPOTS, DIM, DIM, DIM, DIM, 0);
    gemm(WSB(O_E2BF), Wv, F(I_CAUS_VB), nullptr, WSF(O_XVF), nullptr, N_SPOTS, DIM, DIM, DIM, DIM, 0);
    transpose_cvt<<<dim3(DIM / 32, N_SPOTS / 32), 256, 0, stream>>>(
        WSF(O_XVF), WSB(O_XVTBF), N_SPOTS, DIM);
    flash_attn<64><<<dim3(N_SPOTS / 64), 128, 0, stream>>>(
        WSB(O_XQBF), WSB(O_XKBF), WSB(O_XVTBF), Ii(I_CMASK), WSF(O_CAUS),
        N_SPOTS, DIM, DIM, N_SPOTS, DIM, 0.125f);
  }

  // ---- hypergraph ----------------------------------------------------------
  {
    dv_kernel<<<16, 256, 0, stream>>>(F(I_H), WSF(O_DV));
    de_kernel<<<2, 256, 0, stream>>>(F(I_H), WSF(O_DE));
    const bf16* Whg = cw(I_HG, 128 * DIM);
    // X = e1 @ hg[0:64] + e2 @ hg[64:128]
    gemm(WSB(O_E2BF), Whg + 64 * DIM, nullptr, nullptr, WSF(O_X), nullptr, N_SPOTS, DIM, DIM, DIM, DIM, 0);
    gemm(WSB(O_E1BF), Whg, nullptr, WSF(O_X), WSF(O_X), nullptr, N_SPOTS, DIM, DIM, DIM, DIM, 0);
    rowscale_cvt<<<N_SPOTS * DIM / 256, 256, 0, stream>>>(WSF(O_X), WSF(O_DV), WSB(O_XSBF), N_SPOTS * DIM);
    gemm(WSB(O_HTBF), WSB(O_XSBF), nullptr, nullptr, WSF(O_T1), nullptr, NEDGE, DIM, N_SPOTS, DIM, DIM, 0);
    rowscale_cvt<<<NEDGE * DIM / 256, 256, 0, stream>>>(WSF(O_T1), WSF(O_DE), WSB(O_T1SBF), NEDGE * DIM);
    gemm(WSB(O_HBF), WSB(O_T1SBF), nullptr, nullptr, WSF(O_T2), nullptr, N_SPOTS, DIM, NEDGE, DIM, DIM, 0);
  }

  // ---- final combine + decoder --------------------------------------------
  final1_k<<<N_SPOTS * DIM / 256, 256, 0, stream>>>(
      WSF(O_E1), WSF(O_CAUS), WSF(O_T2), WSF(O_DV), WSB(O_FIN1BF), N_SPOTS * DIM);
  final2_k<<<N_SPOTS * DIM / 256, 256, 0, stream>>>(
      WSF(O_E2), WSF(O_CATT), WSB(O_FIN2BF), N_SPOTS * DIM);

  const bf16* Wd1 = cw(I_DEC1, (size_t)DIM * DIN1_C);
  const bf16* Wd2 = cw(I_DEC2, (size_t)DIM * DIN2_C);
  gemm(WSB(O_FIN1BF), Wd1, nullptr, nullptr, nullptr, WSB(O_Y1BF), N_SPOTS, DIN1_C, DIM, DIN1_C, DIN1_C, 0);
  gemm(WSB(O_FIN2BF), Wd2, nullptr, nullptr, nullptr, WSB(O_Y2BF), N_SPOTS, DIN2_C, DIM, DIN2_C, DIN2_C, 0);

  float* outp = (float*)d_out;  // [N, 1280]
  gemm(WSB(O_ADJBF), WSB(O_Y1BF), nullptr, nullptr, outp, nullptr,
       N_SPOTS, DIN1_C, N_SPOTS, DIN1_C, DIN1_C + DIN2_C, 1);
  gemm(WSB(O_ADJBF), WSB(O_Y2BF), nullptr, nullptr, outp + DIN1_C, nullptr,
       N_SPOTS, DIN2_C, N_SPOTS, DIN2_C, DIN1_C + DIN2_C, 1);
}
